// MultiHeadAttention_13924283974341
// MI455X (gfx1250) — matrix operand order repaired and checked by reading, not compiled, not run on hardware
//
#include <hip/hip_runtime.h>
#include <hip/hip_bf16.h>

typedef __bf16 bf16_t;
typedef __attribute__((ext_vector_type(8)))  __bf16 v8bf;
typedef __attribute__((ext_vector_type(16))) __bf16 v16bf;
typedef __attribute__((ext_vector_type(8)))  float  v8f;

#define WMMA_BF16(a, b, c) \
  __builtin_amdgcn_wmma_f32_16x16x32_bf16(false, (a), false, (b), (short)0, (c), false, false)

union AFragU { v16bf v; v8bf h[2]; };

// Build a 16-element bf16 fragment from two 16-byte (8 x bf16) chunks.
// All call sites guarantee 16B alignment of p0/p1.
static __device__ __forceinline__ v16bf load_frag(const __bf16* p0, const __bf16* p1) {
  AFragU u;
  u.h[0] = *(const v8bf*)p0;
  u.h[1] = *(const v8bf*)p1;
  return u.v;
}

static __device__ __forceinline__ v8f vzero8() {
  v8f z = {0.f, 0.f, 0.f, 0.f, 0.f, 0.f, 0.f, 0.f};
  return z;
}

// ---------------------------------------------------------------------------
// Kernel 1: per-head Q/K/V projections.
//   Q[b,h,s,k] = sum_d x[b,s,d] * W_q[h,d,k]   (likewise K, V)
// grid = (S/128, 3*H, B), block = 256 (8 waves). Wave w owns 16 rows, 4 N-tiles.
// Outputs bf16: Qb,Kb as [bh, s, 64]; Vt transposed as [bh, dv, s].
// ---------------------------------------------------------------------------
__global__ __launch_bounds__(256) void qkv_proj_kernel(
    const float* __restrict__ x,  const float* __restrict__ Wq,
    const float* __restrict__ Wk, const float* __restrict__ Wv,
    bf16_t* __restrict__ Qb, bf16_t* __restrict__ Kb, bf16_t* __restrict__ Vt)
{
  constexpr int S = 2048, D = 1024, H = 16;
  __shared__ __bf16 xs[128 * 40];   // 128 rows of x-tile, K-chunk 32, pitch 40
  __shared__ __bf16 wsT[64 * 40];   // W tile transposed: [n=64][k=32], pitch 40

  const int tid  = threadIdx.x;
  const int w    = tid >> 5;
  const int l    = tid & 31;
  const int lo   = l & 15;
  const int half = l >> 4;

  const int m0    = blockIdx.x * 128;
  const int which = (int)blockIdx.y % 3;   // 0=Q 1=K 2=V
  const int h     = (int)blockIdx.y / 3;
  const int b     = (int)blockIdx.z;

  const float* W  = (which == 0 ? Wq : (which == 1 ? Wk : Wv)) + (size_t)h * D * 64;
  const float* xb = x + (size_t)b * S * D;

  v8f acc[4];
  #pragma unroll
  for (int nt = 0; nt < 4; ++nt) acc[nt] = vzero8();

  for (int d0 = 0; d0 < D; d0 += 32) {
    // stage x tile (fp32 -> bf16), coalesced
    for (int idx = tid; idx < 128 * 32; idx += 256) {
      int r = idx >> 5, c = idx & 31;
      xs[r * 40 + c] = (__bf16)xb[(size_t)(m0 + r) * D + d0 + c];
    }
    // stage W tile transposed (fp32 -> bf16)
    {
      int r = tid >> 3, cs = (tid & 7) * 8;
      #pragma unroll
      for (int j = 0; j < 8; ++j)
        wsT[(cs + j) * 40 + r] = (__bf16)W[(size_t)(d0 + r) * 64 + cs + j];
    }
    __syncthreads();

    const __bf16* arow = &xs[(w * 16 + lo) * 40];
    v16bf a = load_frag(arow + half * 16, arow + half * 16 + 8);
    #pragma unroll
    for (int nt = 0; nt < 4; ++nt) {
      const __bf16* brow = &wsT[(nt * 16 + lo) * 40 + half * 16];
      v16bf bfrag = load_frag(brow, brow + 8);
      acc[nt] = WMMA_BF16(a, bfrag, acc[nt]);
    }
    __syncthreads();
  }

  // Epilogue: convert once, then one uniform branch on `which` with clean
  // unrolled store loops (constant offsets folded into the store immediates).
  __bf16 res[4][8];
  #pragma unroll
  for (int nt = 0; nt < 4; ++nt)
    #pragma unroll
    for (int r = 0; r < 8; ++r) res[nt][r] = (__bf16)acc[nt][r];

  const size_t bh   = (size_t)b * H + h;
  const int    srow = m0 + w * 16 + 8 * half;     // first of 8 consecutive rows

  if (which == 2) {
    // Vt[bh, col, s] : per lane, fixed s-column stride, 4 col-tiles x 8 rows
    bf16_t* dst = Vt + (bh * 64 + lo) * S + srow;
    #pragma unroll
    for (int nt = 0; nt < 4; ++nt) {
      bf16_t* d = dst + (size_t)nt * 16 * S;
      #pragma unroll
      for (int r = 0; r < 8; ++r) d[r] = res[nt][r];
    }
  } else {
    bf16_t* dst = (which == 1 ? Kb : Qb) + (bh * S + srow) * 64 + lo;
    #pragma unroll
    for (int nt = 0; nt < 4; ++nt) {
      bf16_t* d = dst + nt * 16;
      #pragma unroll
      for (int r = 0; r < 8; ++r) d[(size_t)r * 64] = res[nt][r];
    }
  }
}

// ---------------------------------------------------------------------------
// Kernel 2: flash attention per (b,h). grid = (S/128, H, B), block = 256.
// Wave w owns 16 query rows; loops keys in chunks of 32 with online softmax.
// ---------------------------------------------------------------------------
__global__ __launch_bounds__(256) void attn_kernel(
    const bf16_t* __restrict__ Qb, const bf16_t* __restrict__ Kb,
    const bf16_t* __restrict__ Vt, bf16_t* __restrict__ Ob)
{
  constexpr int S = 2048, H = 16;
  __shared__ __bf16 Ps[8 * 16 * 40];   // per-wave 16x32 P tile, pitch 40

  const int tid  = threadIdx.x;
  const int w    = tid >> 5;
  const int l    = tid & 31;
  const int lo   = l & 15;
  const int half = l >> 4;

  const int b = (int)blockIdx.z, h = (int)blockIdx.y;
  const size_t bh = (size_t)b * H + h;
  const __bf16* Qh = Qb + bh * (size_t)S * 64;
  const __bf16* Kh = Kb + bh * (size_t)S * 64;
  const __bf16* Vh = Vt + bh * (size_t)64 * S;
  __bf16* Pw = &Ps[w * 640];

  const int qrow = (int)blockIdx.x * 128 + w * 16 + lo;
  const __bf16* qp = &Qh[(size_t)qrow * 64];
  v16bf aq0 = load_frag(qp +      half * 16, qp +  8 + half * 16);   // dk 0..31
  v16bf aq1 = load_frag(qp + 32 + half * 16, qp + 40 + half * 16);   // dk 32..63

  float m[8], lsum[8];
  v8f oacc[4];
  #pragma unroll
  for (int r = 0; r < 8; ++r) { m[r] = -3.0e38f; lsum[r] = 0.f; }
  #pragma unroll
  for (int nt = 0; nt < 4; ++nt) oacc[nt] = vzero8();

  for (int t0 = 0; t0 < S; t0 += 32) {
    if (t0 + 32 < S)
      __builtin_prefetch((const void*)&Kh[(size_t)(t0 + 32 + lo) * 64], 0, 3);

    // scores: 16 queries x 32 keys, contraction dk=64 (two WMMA K-steps)
    v8f s0 = vzero8(), s1 = vzero8();
    {
      const __bf16* k0 = &Kh[(size_t)(t0 + lo) * 64 + half * 16];
      const __bf16* k1 = &Kh[(size_t)(t0 + 16 + lo) * 64 + half * 16];
      v16bf bk;
      bk = load_frag(k0,      k0 + 8);  s0 = WMMA_BF16(aq0, bk, s0);
      bk = load_frag(k0 + 32, k0 + 40); s0 = WMMA_BF16(aq1, bk, s0);
      bk = load_frag(k1,      k1 + 8);  s1 = WMMA_BF16(aq0, bk, s1);
      bk = load_frag(k1 + 32, k1 + 40); s1 = WMMA_BF16(aq1, bk, s1);
    }

    // online softmax (rows = r + 8*half; masks 1/2/4/8 stay within 16-lane halves)
    float alpha[8];
    #pragma unroll
    for (int r = 0; r < 8; ++r) {
      float a0 = s0[r] * 0.125f;           // 1/sqrt(64)
      float a1 = s1[r] * 0.125f;
      float mx = fmaxf(a0, a1);
      mx = fmaxf(mx, __shfl_xor(mx, 1, 32));
      mx = fmaxf(mx, __shfl_xor(mx, 2, 32));
      mx = fmaxf(mx, __shfl_xor(mx, 4, 32));
      mx = fmaxf(mx, __shfl_xor(mx, 8, 32));
      float mn = fmaxf(m[r], mx);
      float al = __expf(m[r] - mn);
      alpha[r] = al; m[r] = mn;
      float p0 = __expf(a0 - mn), p1 = __expf(a1 - mn);
      float ps = p0 + p1;
      ps += __shfl_xor(ps, 1, 32);
      ps += __shfl_xor(ps, 2, 32);
      ps += __shfl_xor(ps, 4, 32);
      ps += __shfl_xor(ps, 8, 32);
      lsum[r] = lsum[r] * al + ps;
      // C-layout -> LDS, row-major P[query][key] (bf16)
      Pw[(r + 8 * half) * 40 + lo]      = (__bf16)p0;
      Pw[(r + 8 * half) * 40 + 16 + lo] = (__bf16)p1;
    }
    #pragma unroll
    for (int nt = 0; nt < 4; ++nt)
      #pragma unroll
      for (int r = 0; r < 8; ++r) oacc[nt][r] *= alpha[r];

    __builtin_amdgcn_wave_barrier();   // order P stores vs A-frag loads (same wave)

    // P (16x32) A-fragment, then P.V with V transposed -> contiguous B-frags
    const __bf16* pr = &Pw[lo * 40];
    v16bf pa = load_frag(pr + half * 16, pr + half * 16 + 8);
    #pragma unroll
    for (int nt = 0; nt < 4; ++nt) {
      const __bf16* vp = &Vh[(size_t)(nt * 16 + lo) * S + t0 + half * 16];
      v16bf bv = load_frag(vp, vp + 8);
      oacc[nt] = WMMA_BF16(pa, bv, oacc[nt]);
    }
    __builtin_amdgcn_wave_barrier();   // order A-frag loads vs next-iter stores
  }

  // epilogue: normalize and store; hoist base address, constant offsets inside
  float inv[8];
  #pragma unroll
  for (int r = 0; r < 8; ++r) inv[r] = 1.0f / lsum[r];

  const int s0row = (int)blockIdx.x * 128 + w * 16 + 8 * half;
  bf16_t* dst = Ob + ((size_t)b * S + s0row) * 1024 + h * 64 + lo;
  #pragma unroll
  for (int nt = 0; nt < 4; ++nt) {
    bf16_t* d = dst + nt * 16;
    #pragma unroll
    for (int r = 0; r < 8; ++r)
      d[(size_t)r * 1024] = (__bf16)(oacc[nt][r] * inv[r]);
  }
}

// ---------------------------------------------------------------------------
// Kernel 3: output projection. out[m,n] = sum_d O[m,d] * Wo[d,n], fp32 out.
// grid = (B*S/128, 1024/64), block = 256.
// ---------------------------------------------------------------------------
__global__ __launch_bounds__(256) void oproj_kernel(
    const bf16_t* __restrict__ Ob, const float* __restrict__ Wo,
    float* __restrict__ out)
{
  constexpr int D = 1024;
  __shared__ __bf16 os[128 * 40];
  __shared__ __bf16 wsT[64 * 40];

  const int tid  = threadIdx.x;
  const int w    = tid >> 5;
  const int l    = tid & 31;
  const int lo   = l & 15;
  const int half = l >> 4;

  const int m0 = (int)blockIdx.x * 128;
  const int n0 = (int)blockIdx.y * 64;

  v8f acc[4];
  #pragma unroll
  for (int nt = 0; nt < 4; ++nt) acc[nt] = vzero8();

  for (int d0 = 0; d0 < D; d0 += 32) {
    for (int idx = tid; idx < 128 * 32; idx += 256) {
      int r = idx >> 5, c = idx & 31;
      os[r * 40 + c] = Ob[(size_t)(m0 + r) * D + d0 + c];
    }
    {
      int r = tid >> 3, cs = (tid & 7) * 8;
      #pragma unroll
      for (int j = 0; j < 8; ++j)
        wsT[(cs + j) * 40 + r] = (__bf16)Wo[(size_t)(d0 + r) * D + n0 + cs + j];
    }
    __syncthreads();

    const __bf16* arow = &os[(w * 16 + lo) * 40];
    v16bf a = load_frag(arow + half * 16, arow + half * 16 + 8);
    #pragma unroll
    for (int nt = 0; nt < 4; ++nt) {
      const __bf16* brow = &wsT[(nt * 16 + lo) * 40 + half * 16];
      v16bf bfrag = load_frag(brow, brow + 8);
      acc[nt] = WMMA_BF16(a, bfrag, acc[nt]);
    }
    __syncthreads();
  }

  const int m0row = m0 + w * 16 + 8 * half;
  float* dst = out + (size_t)m0row * D + n0 + lo;
  #pragma unroll
  for (int nt = 0; nt < 4; ++nt) {
    float* d = dst + nt * 16;
    #pragma unroll
    for (int r = 0; r < 8; ++r) d[(size_t)r * D] = acc[nt][r];
  }
}

// ---------------------------------------------------------------------------
extern "C" void kernel_launch(void* const* d_in, const int* in_sizes, int n_in,
                              void* d_out, int out_size, void* d_ws, size_t ws_size,
                              hipStream_t stream) {
  (void)in_sizes; (void)n_in; (void)out_size; (void)ws_size;
  const float* x  = (const float*)d_in[0];
  const float* Wq = (const float*)d_in[1];
  const float* Wk = (const float*)d_in[2];
  const float* Wv = (const float*)d_in[3];
  const float* Wo = (const float*)d_in[4];
  float* out = (float*)d_out;

  const size_t per = (size_t)4 * 16 * 2048 * 64;  // B*H*S*64 bf16 elements
  bf16_t* Qb = (bf16_t*)d_ws;
  bf16_t* Kb = Qb + per;
  bf16_t* Vt = Kb + per;
  bf16_t* Ob = Vt + per;

  qkv_proj_kernel<<<dim3(16, 48, 4), 256, 0, stream>>>(x, Wq, Wk, Wv, Qb, Kb, Vt);
  attn_kernel   <<<dim3(16, 16, 4), 256, 0, stream>>>(Qb, Kb, Vt, Ob);
  oproj_kernel  <<<dim3(64, 16, 1), 256, 0, stream>>>(Ob, Wo, out);
}
